// QRNN_62139586838571
// MI455X (gfx1250) — matrix operand order repaired and checked by reading, not compiled, not run on hardware
//
#include <hip/hip_runtime.h>
#include <stdint.h>

#define B_   8
#define T_   1024
#define H_   1024
#define G3H  3072

#define MT   128
#define NT   64
#define KT   32
#define ASTR 40   // padded LDS stride (bf16 elems): 80B, 16B-aligned, conflict-free
#define BSTR 40
#define ABUF ((MT + 2) * ASTR)   // one A buffer (elems)
#define BBUF (3 * NT * BSTR)     // one B buffer (elems)

#ifndef HAVE_ASYNC
#if defined(__has_builtin)
#if __has_builtin(__builtin_amdgcn_global_load_async_to_lds_b128)
#define HAVE_ASYNC 1
#endif
#endif
#endif
#ifndef HAVE_ASYNC
#define HAVE_ASYNC 0
#endif

typedef __attribute__((ext_vector_type(16))) __bf16 v16bf;
typedef __attribute__((ext_vector_type(8)))  float  v8f;
typedef int async_v4i __attribute__((vector_size(16)));   // builtin's pointee type
typedef unsigned int u32x4 __attribute__((vector_size(16)));

__device__ __forceinline__ unsigned short f2bf(float x) {
  union { float f; unsigned int u; } c; c.f = x;
  unsigned int u = c.u;
  unsigned int r = u + 0x7FFFu + ((u >> 16) & 1u);   // round-to-nearest-even
  return (unsigned short)(r >> 16);
}

// 16-byte global -> LDS copy. Async path (ASYNCcnt) when available,
// else synchronous load+store.
__device__ __forceinline__ void copy16_g2l(unsigned short* dst,
                                           const unsigned short* src) {
#if HAVE_ASYNC
  __builtin_amdgcn_global_load_async_to_lds_b128(
      (async_v4i*)src, (async_v4i*)dst, 0, 0);
#else
  *(u32x4*)dst = *(const u32x4*)src;
#endif
}

__device__ __forceinline__ void zero16_l(unsigned short* dst) {
  *(u32x4*)dst = (u32x4){0u, 0u, 0u, 0u};
}

__device__ __forceinline__ void wait_async0() {
#if HAVE_ASYNC
#if defined(__has_builtin) && __has_builtin(__builtin_amdgcn_s_wait_asynccnt)
  __builtin_amdgcn_s_wait_asynccnt(0);
#else
  asm volatile("s_wait_asynccnt 0" ::: "memory");
#endif
#endif
}

__global__ void cvt_f32_bf16(const float* __restrict__ in,
                             unsigned short* __restrict__ out, int n) {
  int i = blockIdx.x * blockDim.x + threadIdx.x;
  if (i < n) out[i] = f2bf(in[i]);
}

// w [3H, C, 3] fp32 -> wbf [3][3H][C] bf16 (row-major per output channel)
__global__ void cvt_w(const float* __restrict__ w,
                      unsigned short* __restrict__ wbf, int C) {
  int i = blockIdx.x * blockDim.x + threadIdx.x;
  int n = G3H * C * 3;
  if (i >= n) return;
  int k  = i % 3;
  int ci = (i / 3) % C;
  int o  = i / (3 * C);
  wbf[((size_t)k * G3H + o) * C + ci] = f2bf(w[i]);
}

// Gates GEMM with fused K=3 conv, bias, activation; double-buffered LDS with
// async global->LDS staging pipelined against WMMA compute.
// g[b,t,o] = sum_k sum_i x[b,t+k-1,i] * W_k[o,i]  ; f,o sigmoid ; z tanh
__global__ __launch_bounds__(256) void gemm_gates(
    const unsigned short* __restrict__ xbf, int C,
    const unsigned short* __restrict__ wbf,
    const float* __restrict__ bias,
    float* __restrict__ fb, float* __restrict__ ob, float* __restrict__ zb)
{
  __shared__ unsigned short Alds[2 * ABUF];   // 2 x (130 halo rows x 32 bf16)
  __shared__ unsigned short Blds[2 * BBUF];   // 2 x (3 kernels x 64 rows x 32 bf16)

  const int tid  = threadIdx.x;
  const int lane = tid & 31;
  const int wid  = tid >> 5;
  const int wm   = wid & 3;     // 4 waves along M
  const int wn   = wid >> 2;    // 2 waves along N

  const int m0    = blockIdx.x * MT;   // row in [0, B*T)
  const int n0    = blockIdx.y * NT;   // col in [0, 3H)
  const int b     = m0 / T_;
  const int tbase = m0 % T_;           // tile fully inside one batch (128 | 1024)

  v8f acc[2][2];
#pragma unroll
  for (int i = 0; i < 2; ++i)
#pragma unroll
    for (int j = 0; j < 2; ++j)
      acc[i][j] = (v8f){0.f, 0.f, 0.f, 0.f, 0.f, 0.f, 0.f, 0.f};

  const int nk = C / KT;

  // ---- tile-load stage (issues per-lane 16B async copies) ----
  auto load_tiles = [&](int c0, int buf) {
    unsigned short* A = &Alds[buf * ABUF];
    unsigned short* Bl = &Blds[buf * BBUF];
    // A halo tile: rows t = tbase-1 .. tbase+128, zero-padded at sequence edges
    for (int idx = tid; idx < (MT + 2) * 4; idx += 256) {
      int hr = idx >> 2;
      int q  = idx & 3;
      int t  = tbase - 1 + hr;
      unsigned short* dst = &A[hr * ASTR + q * 8];
      if (t >= 0 && t < T_)
        copy16_g2l(dst, xbf + ((size_t)(b * T_ + t) * C + c0 + q * 8));
      else
        zero16_l(dst);
    }
    // B tiles: W_k rows (output-channel major => K-contiguous per column)
    for (int idx = tid; idx < 3 * NT * 4; idx += 256) {
      int row = idx >> 2;       // 0..191
      int q   = idx & 3;
      int k   = row / NT;
      int nl  = row % NT;
      copy16_g2l(&Bl[(k * NT + nl) * BSTR + q * 8],
                 wbf + ((size_t)(k * G3H + n0 + nl) * C + c0 + q * 8));
    }
  };

  load_tiles(0, 0);   // prologue

  for (int kc = 0; kc < nk; ++kc) {
    const int cur = kc & 1;
    wait_async0();          // this wave's staged copies for `cur` are in LDS
    __syncthreads();        // all waves' copies visible; prior reads of `cur^1` done
    if (kc + 1 < nk)
      load_tiles((kc + 1) * KT, cur ^ 1);   // overlap next tile with compute

    const unsigned short* A = &Alds[cur * ABUF];
    const unsigned short* Bl = &Blds[cur * BBUF];
    const int kb = (lane >> 4) * 8;    // A: K-half select
    const int k2 = (lane >> 4) * 8;    // B: K-half select
    const int mr = lane & 15;          // A row / B column within fragment
#pragma unroll
    for (int s = 0; s < 3; ++s) {
      union { u32x4 u[2]; v16bf v; } afrag[2], bfrag[2];
#pragma unroll
      for (int i = 0; i < 2; ++i) {
        const unsigned short* ap = &A[(wm * 32 + i * 16 + mr + s) * ASTR + kb];
        afrag[i].u[0] = *(const u32x4*)ap;          // K = kb+0..7
        afrag[i].u[1] = *(const u32x4*)(ap + 16);   // K = 16+kb+0..7
      }
#pragma unroll
      for (int j = 0; j < 2; ++j) {
        const unsigned short* bp =
            &Bl[(s * NT + wn * 32 + j * 16 + mr) * BSTR + k2];
        bfrag[j].u[0] = *(const u32x4*)bp;          // K = k2+0..7
        bfrag[j].u[1] = *(const u32x4*)(bp + 16);   // K = 16+k2+0..7
      }
#pragma unroll
      for (int i = 0; i < 2; ++i)
#pragma unroll
        for (int j = 0; j < 2; ++j)
          acc[i][j] = __builtin_amdgcn_wmma_f32_16x16x32_bf16(
              false, afrag[i].v, false, bfrag[j].v, (short)0, acc[i][j],
              false, false);
    }
  }

  // Epilogue: bias + activation, scatter to f/o/z planes
  const int lhi = lane >> 4;
#pragma unroll
  for (int i = 0; i < 2; ++i) {
#pragma unroll
    for (int j = 0; j < 2; ++j) {
      int n_g   = n0 + wn * 32 + j * 16 + (lane & 15);
      float bv  = bias[n_g];
      int plane = n_g >> 10;            // 0=f, 1=o, 2=z (16-wide frag never crosses)
      int h     = n_g & (H_ - 1);
      float* dst = (plane == 0) ? fb : ((plane == 1) ? ob : zb);
#pragma unroll
      for (int v = 0; v < 8; ++v) {
        int t   = tbase + wm * 32 + i * 16 + lhi * 8 + v;  // C-layout: M = v + 8*(lane>>4)
        float g = acc[i][j][v] + bv;
        float a = (plane == 2) ? tanhf(g) : 1.0f / (1.0f + __expf(-g));
        dst[(size_t)(b * T_ + t) * H_ + h] = a;
      }
    }
  }
}

// fo-pool scan: c_t = f*c + (1-f)*z ; h = c*o. One thread per (b,h).
__global__ void fo_scan(const float* __restrict__ f, const float* __restrict__ o,
                        const float* __restrict__ z, int dir, int layer,
                        unsigned short* __restrict__ hbf,  // [B,T,2H] bf16 or null
                        float* __restrict__ hf32,          // [B,T,2H] f32 or null
                        float* __restrict__ hid, float* __restrict__ cell)
{
  int tidg = blockIdx.x * blockDim.x + threadIdx.x;
  if (tidg >= B_ * H_) return;
  int b = tidg >> 10;
  int h = tidg & (H_ - 1);
  float c = 0.f;
  for (int s = 0; s < T_; ++s) {
    int t = dir ? (T_ - 1 - s) : s;
    size_t idx = (size_t)(b * T_ + t) * H_ + h;
    float ft = f[idx];
    float zt = z[idx];
    c = ft * c + (1.f - ft) * zt;
    float hv = c * o[idx];
    size_t oidx = (size_t)(b * T_ + t) * (2 * H_) + (size_t)dir * H_ + h;
    if (hbf)  hbf[oidx]  = f2bf(hv);
    if (hf32) hf32[oidx] = hv;
    if (t == T_ - 1) {    // last_h / last_c live at t=T-1 for both directions
      int hi = layer * 16384 + b * 2048 + dir * 1024 + h;
      hid[hi]  = hv;
      cell[hi] = c;
    }
  }
}

extern "C" void kernel_launch(void* const* d_in, const int* in_sizes, int n_in,
                              void* d_out, int out_size, void* d_ws, size_t ws_size,
                              hipStream_t stream) {
  (void)in_sizes; (void)n_in; (void)out_size; (void)ws_size;
  const float* x   = (const float*)d_in[0];
  // d_in[1] = lengths (unused; all T)
  const float* w0f = (const float*)d_in[2];
  const float* b0f = (const float*)d_in[3];
  const float* w0r = (const float*)d_in[4];
  const float* b0r = (const float*)d_in[5];
  const float* w1f = (const float*)d_in[6];
  const float* b1f = (const float*)d_in[7];
  const float* w1r = (const float*)d_in[8];
  const float* b1r = (const float*)d_in[9];

  float* out1 = (float*)d_out;                         // [B,T,2H]
  float* hid  = out1 + (size_t)B_ * T_ * 2 * H_;       // [4,B,H]
  float* cell = hid + 4 * B_ * H_;                     // [4,B,H]

  char* ws = (char*)d_ws;
  unsigned short* xbf0 = (unsigned short*)ws; ws += (size_t)B_ * T_ * 1024 * 2;
  unsigned short* xbf1 = (unsigned short*)ws; ws += (size_t)B_ * T_ * 2048 * 2;
  unsigned short* wbf  = (unsigned short*)ws; ws += (size_t)3 * G3H * 2048 * 2;
  float* fb = (float*)ws; ws += (size_t)B_ * T_ * H_ * 4;
  float* ob = (float*)ws; ws += (size_t)B_ * T_ * H_ * 4;
  float* zb = (float*)ws; ws += (size_t)B_ * T_ * H_ * 4;

  dim3 gg(64, 48);   // (B*T/128, 3H/64)

  int n = B_ * T_ * 1024;
  cvt_f32_bf16<<<(n + 255) / 256, 256, 0, stream>>>(x, xbf0, n);

  // ---- Layer 0 (C=1024): writes bf16 concat input for layer 1 ----
  const float* w0[2] = {w0f, w0r};
  const float* bb0[2] = {b0f, b0r};
  for (int dir = 0; dir < 2; ++dir) {
    int nw = G3H * 1024 * 3;
    cvt_w<<<(nw + 255) / 256, 256, 0, stream>>>(w0[dir], wbf, 1024);
    gemm_gates<<<gg, 256, 0, stream>>>(xbf0, 1024, wbf, bb0[dir], fb, ob, zb);
    fo_scan<<<(B_ * H_) / 256, 256, 0, stream>>>(fb, ob, zb, dir, 0,
                                                 xbf1, nullptr, hid, cell);
  }

  // ---- Layer 1 (C=2048): writes final f32 output ----
  const float* w1[2] = {w1f, w1r};
  const float* bb1[2] = {b1f, b1r};
  for (int dir = 0; dir < 2; ++dir) {
    int nw = G3H * 2048 * 3;
    cvt_w<<<(nw + 255) / 256, 256, 0, stream>>>(w1[dir], wbf, 2048);
    gemm_gates<<<gg, 256, 0, stream>>>(xbf1, 2048, wbf, bb1[dir], fb, ob, zb);
    fo_scan<<<(B_ * H_) / 256, 256, 0, stream>>>(fb, ob, zb, dir, 1,
                                                 nullptr, out1, hid, cell);
  }
}
